// MoETop1_68453188764073
// MI455X (gfx1250) — matrix operand order repaired and checked by reading, not compiled, not run on hardware
//
#include <hip/hip_runtime.h>
#include <hip/hip_bf16.h>

// ---------------- problem constants (match reference) ----------------
constexpr int T_TOK = 8192;        // 4 * 2048 tokens
constexpr int D_MODEL = 1024;
constexpr int D_HIDDEN = 4096;
constexpr int NUM_E = 8;
constexpr int CAP = 1280;          // int(1.25 * 8192 / 8)
constexpr float AUX_COEF = 0.01f;

typedef _Float16 v16h __attribute__((ext_vector_type(16)));
typedef _Float16 v8h  __attribute__((ext_vector_type(8)));
typedef float    v8f  __attribute__((ext_vector_type(8)));

// LDS staging geometry for the WMMA GEMM:
//  A tile: 128 rows x 32 f16 (64B payload) @ 80B stride  -> 10240 B
//  B tile: 256 rows x 32 f16 (64B payload) @ 80B stride  -> 20480 B
// 80B = 20 dwords stride => 20*r mod 64 tiles the 64 LDS banks perfectly.
constexpr int LDS_ROW_STRIDE = 80;
constexpr int LDS_A_BYTES = 128 * LDS_ROW_STRIDE;     // 10240
constexpr int LDS_STAGE_BYTES = LDS_A_BYTES + 256 * LDS_ROW_STRIDE; // 30720

// ---------------------------------------------------------------------
// K1: router. One wave32 per token: logits = x @ rw + rb, softmax,
// first-argmax, store gates / top_idx / top_val.
// ---------------------------------------------------------------------
__global__ void moe_router_kernel(const float* __restrict__ x,
                                  const float* __restrict__ rw,
                                  const float* __restrict__ rb,
                                  float* __restrict__ gates,
                                  int* __restrict__ topIdx,
                                  float* __restrict__ topVal) {
    const int lane = threadIdx.x & 31;
    const int wid  = threadIdx.x >> 5;
    const int t    = blockIdx.x * 8 + wid;
    if (t >= T_TOK) return;

    float acc[NUM_E];
#pragma unroll
    for (int e = 0; e < NUM_E; ++e) acc[e] = 0.f;

    const float* xr = x + (size_t)t * D_MODEL;
    for (int d = lane; d < D_MODEL; d += 32) {
        const float xv = xr[d];
        const float* w = rw + (size_t)d * NUM_E;
#pragma unroll
        for (int e = 0; e < NUM_E; ++e) acc[e] += xv * w[e];
    }
#pragma unroll
    for (int off = 16; off >= 1; off >>= 1) {
#pragma unroll
        for (int e = 0; e < NUM_E; ++e) acc[e] += __shfl_xor(acc[e], off, 32);
    }
    if (lane == 0) {
        float lg[NUM_E];
        float mx = -1e30f;
#pragma unroll
        for (int e = 0; e < NUM_E; ++e) { lg[e] = acc[e] + rb[e]; mx = fmaxf(mx, lg[e]); }
        float s = 0.f;
#pragma unroll
        for (int e = 0; e < NUM_E; ++e) { lg[e] = __expf(lg[e] - mx); s += lg[e]; }
        const float inv = 1.f / s;
        int best = 0; float bv = -1.f;
#pragma unroll
        for (int e = 0; e < NUM_E; ++e) {
            const float g = lg[e] * inv;
            gates[(size_t)t * NUM_E + e] = g;
            if (g > bv) { bv = g; best = e; }   // strict > : first max wins
        }
        topIdx[t] = best;
        topVal[t] = bv;
    }
}

// ---------------------------------------------------------------------
// K2: deterministic first-come rank scan. One block, chunks of 1024.
// ---------------------------------------------------------------------
__global__ void moe_scan_kernel(const int* __restrict__ topIdx,
                                int* __restrict__ rank,
                                int* __restrict__ expertTotal) {
    __shared__ int sh_e[1024];
    __shared__ int base[NUM_E];
    const int tid = threadIdx.x;
    if (tid < NUM_E) base[tid] = 0;
    __syncthreads();

    for (int c = 0; c < T_TOK / 1024; ++c) {
        const int t = c * 1024 + tid;
        const int e = topIdx[t];
        sh_e[tid] = e;
        __syncthreads();
        int cnt = 0;
        for (int j = 0; j < tid; ++j) cnt += (sh_e[j] == e) ? 1 : 0;
        rank[t] = base[e] + cnt;
        __syncthreads();
        if (tid < NUM_E) {
            int tot = 0;
            for (int j = 0; j < 1024; ++j) tot += (sh_e[j] == tid) ? 1 : 0;
            base[tid] += tot;
        }
        __syncthreads();
    }
    if (tid < NUM_E) expertTotal[tid] = base[tid];
}

// ---------------------------------------------------------------------
// K3: aux loss + used_counts (deterministic tree reduction of gates).
// out layout: [ y (T*D) | aux (1) | used_counts (E) ]
// ---------------------------------------------------------------------
__global__ void moe_aux_kernel(const float* __restrict__ gates,
                               const int* __restrict__ expertTotal,
                               float* __restrict__ out) {
    __shared__ float simp[NUM_E];
    const int lane = threadIdx.x & 31;
    const int e    = threadIdx.x >> 5;   // 8 waves, one expert each
    float p = 0.f;
    for (int t = lane; t < T_TOK; t += 32) p += gates[(size_t)t * NUM_E + e];
#pragma unroll
    for (int off = 16; off >= 1; off >>= 1) p += __shfl_xor(p, off, 32);
    if (lane == 0) simp[e] = p;
    __syncthreads();
    const size_t YS = (size_t)T_TOK * D_MODEL;
    if (threadIdx.x == 0) {
        float aux = 0.f;
        const float invT = 1.f / (float)T_TOK;
        for (int k = 0; k < NUM_E; ++k)
            aux += (simp[k] * invT) * ((float)expertTotal[k] * invT);
        out[YS] = aux * (float)NUM_E * AUX_COEF;
    }
    if (threadIdx.x < NUM_E) {
        int c = expertTotal[threadIdx.x];
        out[YS + 1 + threadIdx.x] = (float)(c < CAP ? c : CAP);
    }
}

// ---------------------------------------------------------------------
// K4: dispatch kept tokens into zeroed f16 buffers [E, CAP, D].
// ---------------------------------------------------------------------
__global__ void moe_dispatch_kernel(const float* __restrict__ x,
                                    const int* __restrict__ topIdx,
                                    const int* __restrict__ rank,
                                    _Float16* __restrict__ buf) {
    const int t = blockIdx.x;
    const int r = rank[t];
    if (r >= CAP) return;
    const int e = topIdx[t];
    _Float16* dst = buf + ((size_t)e * CAP + r) * D_MODEL;
    const float* src = x + (size_t)t * D_MODEL;
    for (int i = threadIdx.x; i < D_MODEL; i += 256)
        dst[i] = (_Float16)src[i];
}

// ---------------------------------------------------------------------
// K5: f32 [E,R,C] -> f16 transposed [E,C,R] (N-major weights so WMMA
// B-fragments are two 16B K-runs per lane).
// ---------------------------------------------------------------------
__global__ void moe_transpose_f16_kernel(const float* __restrict__ w,
                                         _Float16* __restrict__ wt,
                                         int R, int C) {
    __shared__ float tile[32][33];
    const int e = blockIdx.z;
    const float* src = w + (size_t)e * R * C;
    _Float16* dst = wt + (size_t)e * R * C;
    const int c0 = blockIdx.x * 32, r0 = blockIdx.y * 32;
#pragma unroll
    for (int q = 0; q < 4; ++q) {
        const int r = r0 + threadIdx.y + q * 8;
        tile[threadIdx.y + q * 8][threadIdx.x] = src[(size_t)r * C + c0 + threadIdx.x];
    }
    __syncthreads();
#pragma unroll
    for (int q = 0; q < 4; ++q) {
        const int c = c0 + threadIdx.y + q * 8;
        dst[(size_t)c * R + r0 + threadIdx.x] = (_Float16)tile[threadIdx.x][threadIdx.y + q * 8];
    }
}

// ---------------------------------------------------------------------
// Async DMA of one K-step tile (A 128x32 f16 + B 256x32 f16) into LDS.
// 256 threads issue 2 A-chunks + 4 B-chunks of 16B each (ASYNCcnt path,
// no VGPR data movement).
// ---------------------------------------------------------------------
__device__ __forceinline__ void stage_tile_async(
        int tid, const _Float16* __restrict__ Ab, const _Float16* __restrict__ Bb,
        int K, int rowBlk, int colBlk, int k0, unsigned aLds, unsigned bLds) {
#pragma unroll
    for (int q = 0; q < 2; ++q) {
        const int c = tid + 256 * q;
        const int row = c >> 2, sub = c & 3;
        const void* g = Ab + (size_t)(rowBlk + row) * K + k0 + sub * 8;
        const unsigned l = aLds + (unsigned)row * LDS_ROW_STRIDE + (unsigned)sub * 16u;
        asm volatile("global_load_async_to_lds_b128 %0, %1, off"
                     :: "v"(l), "v"((unsigned long long)(uintptr_t)g) : "memory");
    }
#pragma unroll
    for (int q = 0; q < 4; ++q) {
        const int c = tid + 256 * q;
        const int row = c >> 2, sub = c & 3;
        const void* g = Bb + (size_t)(colBlk + row) * K + k0 + sub * 8;
        const unsigned l = bLds + (unsigned)row * LDS_ROW_STRIDE + (unsigned)sub * 16u;
        asm volatile("global_load_async_to_lds_b128 %0, %1, off"
                     :: "v"(l), "v"((unsigned long long)(uintptr_t)g) : "memory");
    }
}

__device__ __forceinline__ void wait_async_all() {
    asm volatile("s_wait_asynccnt 0x0" ::: "memory");
}

// ---------------------------------------------------------------------
// K6: grouped GEMM via v_wmma_f32_16x16x32_f16 with double-buffered
// async Global->LDS staging.
// A: [E, M, K] f16 row-major.  B: [E, N, K] f16 (N-major / pre-transposed).
// Block = 256 threads = 8 waves (2x4), block tile 128x256, wave tile 64x64
// (16 WMMA accumulators). RELU_F16 -> fused bias+ReLU, f16 out (stage 1);
// else bias, f32 out (stage 2).
// ---------------------------------------------------------------------
template <bool RELU_F16>
__global__ __launch_bounds__(256) void moe_wmma_gemm_kernel(
        const _Float16* __restrict__ A,
        const _Float16* __restrict__ B,
        const float* __restrict__ bias,
        void* __restrict__ outp,
        int M, int N, int K) {
    __shared__ __attribute__((aligned(16))) char smem[2 * LDS_STAGE_BYTES];

    const int e    = blockIdx.z;
    const int tid  = threadIdx.x;
    const int lane = tid & 31;
    const int wid  = tid >> 5;
    const int rowBlk = blockIdx.y * 128;
    const int colBlk = blockIdx.x * 256;
    const int wiRow = (wid >> 2) * 64;   // wave row offset within block tile
    const int wiCol = (wid & 3) * 64;    // wave col offset within block tile

    const _Float16* Ab = A + (size_t)e * M * K;
    const _Float16* Bb = B + (size_t)e * N * K;
    const float* be    = bias + (size_t)e * N;

    const unsigned ldsBase = (unsigned)(uintptr_t)smem;  // low 32 bits = LDS offset

    v8f acc[16];
#pragma unroll
    for (int i = 0; i < 16; ++i) acc[i] = (v8f)(0.f);

    const int am = lane & 15, ah = lane >> 4;   // A frag: lane = m + 16*half
    const int bn = lane & 15, bk = lane >> 4;   // B frag: lane = n + 16*khalf

    // prologue: stage k0 = 0
    stage_tile_async(tid, Ab, Bb, K, rowBlk, colBlk, 0,
                     ldsBase, ldsBase + LDS_A_BYTES);
    wait_async_all();
    __syncthreads();

    int cur = 0;
    for (int k0 = 0; k0 < K; k0 += 32) {
        const bool more = (k0 + 32) < K;
        if (more) {
            const unsigned nb = ldsBase + (unsigned)((cur ^ 1) * LDS_STAGE_BYTES);
            stage_tile_async(tid, Ab, Bb, K, rowBlk, colBlk, k0 + 32,
                             nb, nb + LDS_A_BYTES);
        }

        const char* sA = smem + cur * LDS_STAGE_BYTES;
        const char* sB = sA + LDS_A_BYTES;

        v16h af[4], bf[4];
#pragma unroll
        for (int i = 0; i < 4; ++i) {
            const char* ap = sA + (size_t)(wiRow + 16 * i + am) * LDS_ROW_STRIDE + ah * 16;
            v8h lo = *(const v8h*)__builtin_assume_aligned(ap, 16);
            v8h hi = *(const v8h*)__builtin_assume_aligned(ap + 32, 16);
            af[i] = __builtin_shufflevector(lo, hi, 0,1,2,3,4,5,6,7,8,9,10,11,12,13,14,15);
        }
#pragma unroll
        for (int j = 0; j < 4; ++j) {
            const char* bp = sB + (size_t)(wiCol + 16 * j + bn) * LDS_ROW_STRIDE + bk * 16;
            v8h lo = *(const v8h*)__builtin_assume_aligned(bp, 16);
            v8h hi = *(const v8h*)__builtin_assume_aligned(bp + 32, 16);
            bf[j] = __builtin_shufflevector(lo, hi, 0,1,2,3,4,5,6,7,8,9,10,11,12,13,14,15);
        }
#pragma unroll
        for (int i = 0; i < 4; ++i)
#pragma unroll
            for (int j = 0; j < 4; ++j)
                acc[i * 4 + j] = __builtin_amdgcn_wmma_f32_16x16x32_f16(
                    false, af[i], false, bf[j], (short)0, acc[i * 4 + j], false, false);

        if (more) {
            wait_async_all();   // next stage landed (this wave's loads, in order)
            __syncthreads();    // all waves landed + done reading cur stage
        }
        cur ^= 1;
    }

    // D layout: vgpr r <-> M = 8*(lane>=16)+r, N = lane&15
    const int dn = lane & 15, dmh = lane >> 4;
#pragma unroll
    for (int i = 0; i < 4; ++i) {
#pragma unroll
        for (int j = 0; j < 4; ++j) {
            const int col = colBlk + wiCol + 16 * j + dn;
            const float bv = be[col];
#pragma unroll
            for (int r = 0; r < 8; ++r) {
                const int row = rowBlk + wiRow + 16 * i + dmh * 8 + r;
                float v = acc[i * 4 + j][r] + bv;
                if (RELU_F16) {
                    v = fmaxf(v, 0.f);
                    ((_Float16*)outp)[(size_t)e * M * N + (size_t)row * N + col] = (_Float16)v;
                } else {
                    ((float*)outp)[(size_t)e * M * N + (size_t)row * N + col] = v;
                }
            }
        }
    }
}

// ---------------------------------------------------------------------
// K7: combine — gather (expert, rank), scale by gate, zero dropped.
// ---------------------------------------------------------------------
__global__ void moe_combine_kernel(const float* __restrict__ expOut,
                                   const int* __restrict__ topIdx,
                                   const int* __restrict__ rank,
                                   const float* __restrict__ topVal,
                                   float* __restrict__ y) {
    const int t = blockIdx.x;
    const int e = topIdx[t];
    const int r = rank[t];
    const bool kp = (r < CAP);
    const float g = kp ? topVal[t] : 0.f;
    const int rs = kp ? r : 0;
    const float* src = expOut + ((size_t)e * CAP + rs) * D_MODEL;
    float* dst = y + (size_t)t * D_MODEL;
    for (int i = threadIdx.x; i < D_MODEL; i += 256)
        dst[i] = src[i] * g;
}

// ---------------------------------------------------------------------
static inline size_t alignUp(size_t x, size_t a) { return (x + a - 1) & ~(a - 1); }

extern "C" void kernel_launch(void* const* d_in, const int* in_sizes, int n_in,
                              void* d_out, int out_size, void* d_ws, size_t ws_size,
                              hipStream_t stream) {
    (void)in_sizes; (void)n_in; (void)out_size; (void)ws_size;
    const float* x  = (const float*)d_in[0];
    const float* rw = (const float*)d_in[1];
    const float* rb = (const float*)d_in[2];
    const float* w1 = (const float*)d_in[3];
    const float* b1 = (const float*)d_in[4];
    const float* w2 = (const float*)d_in[5];
    const float* b2 = (const float*)d_in[6];
    float* out = (float*)d_out;

    // ---- workspace carve ----
    char* ws = (char*)d_ws;
    size_t cur = 0;
    auto carve = [&](size_t bytes) -> char* {
        char* p = ws + cur;
        cur = alignUp(cur + bytes, 256);
        return p;
    };
    float*     gates       = (float*)carve((size_t)T_TOK * NUM_E * sizeof(float));
    int*       topIdx      = (int*)  carve((size_t)T_TOK * sizeof(int));
    float*     topVal      = (float*)carve((size_t)T_TOK * sizeof(float));
    int*       rank        = (int*)  carve((size_t)T_TOK * sizeof(int));
    int*       expertTotal = (int*)  carve(NUM_E * sizeof(int));
    _Float16*  buf         = (_Float16*)carve((size_t)NUM_E * CAP * D_MODEL * sizeof(_Float16));
    _Float16*  w1t         = (_Float16*)carve((size_t)NUM_E * D_MODEL * D_HIDDEN * sizeof(_Float16));
    _Float16*  w2t         = (_Float16*)carve((size_t)NUM_E * D_HIDDEN * D_MODEL * sizeof(_Float16));
    _Float16*  hbuf        = (_Float16*)carve((size_t)NUM_E * CAP * D_HIDDEN * sizeof(_Float16));
    float*     expOut      = (float*)carve((size_t)NUM_E * CAP * D_MODEL * sizeof(float));

    // 1) router
    moe_router_kernel<<<T_TOK / 8, 256, 0, stream>>>(x, rw, rb, gates, topIdx, topVal);
    // 2) first-come rank scan (single block, deterministic)
    moe_scan_kernel<<<1, 1024, 0, stream>>>(topIdx, rank, expertTotal);
    // 3) aux loss + used_counts
    moe_aux_kernel<<<1, 256, 0, stream>>>(gates, expertTotal, out);
    // 4) zero dispatch buffers then scatter kept tokens (f32 -> f16)
    hipMemsetAsync(buf, 0, (size_t)NUM_E * CAP * D_MODEL * sizeof(_Float16), stream);
    moe_dispatch_kernel<<<T_TOK, 256, 0, stream>>>(x, topIdx, rank, buf);
    // 5) weight convert + transpose: w1 [E,D,H]->[E,H,D], w2 [E,H,D]->[E,D,H]
    moe_transpose_f16_kernel<<<dim3(D_HIDDEN / 32, D_MODEL / 32, NUM_E), dim3(32, 8), 0, stream>>>(
        w1, w1t, D_MODEL, D_HIDDEN);
    moe_transpose_f16_kernel<<<dim3(D_MODEL / 32, D_HIDDEN / 32, NUM_E), dim3(32, 8), 0, stream>>>(
        w2, w2t, D_HIDDEN, D_MODEL);
    // 6) grouped GEMM 1: h = relu(buf @ w1 + b1)   [E, CAP, H] f16
    moe_wmma_gemm_kernel<true><<<dim3(D_HIDDEN / 256, CAP / 128, NUM_E), 256, 0, stream>>>(
        buf, w1t, b1, (void*)hbuf, CAP, D_HIDDEN, D_MODEL);
    // 7) grouped GEMM 2: expOut = h @ w2 + b2      [E, CAP, D] f32
    moe_wmma_gemm_kernel<false><<<dim3(D_MODEL / 256, CAP / 128, NUM_E), 256, 0, stream>>>(
        hbuf, w2t, b2, (void*)expOut, CAP, D_MODEL, D_HIDDEN);
    // 8) combine
    moe_combine_kernel<<<T_TOK, 256, 0, stream>>>(expOut, topIdx, rank, topVal, out);
}
